// GATModel_17076789969391
// MI455X (gfx1250) — matrix operand order repaired and checked by reading, not compiled, not run on hardware
//
#include <hip/hip_runtime.h>
#include <stdint.h>

// ---- problem constants (match reference) ----
#define NNODES 50000
#define NEDGES 800000
#define EPRIME (NEDGES + NNODES)   // with self loops
#define DIN    128
#define NH     4
#define CC     64
#define NG     512
#define NOUTC  10
#define BN_EPS 1e-5f
#define NEG_SLOPE 0.2f

// ---- CDNA5 WMMA types ----
typedef __attribute__((ext_vector_type(16))) __bf16 v16bf;
typedef __attribute__((ext_vector_type(8)))  float  v8f;

union BF16x16 { v16bf v; unsigned short u[16]; unsigned int w[8]; };

__device__ __forceinline__ unsigned short f2bf(float f) {
    unsigned int u = __float_as_uint(f);
    u += 0x7fffu + ((u >> 16) & 1u);      // round-to-nearest-even
    return (unsigned short)(u >> 16);
}

// monotone float <-> uint key for atomicMax on floats
__device__ __forceinline__ unsigned int fkey(float x) {
    unsigned int b = __float_as_uint(x);
    return (b & 0x80000000u) ? ~b : (b | 0x80000000u);
}
__device__ __forceinline__ float fkeyinv(unsigned int k) {
    unsigned int b = (k & 0x80000000u) ? (k & 0x7fffffffu) : ~k;
    return __uint_as_float(b);
}

__device__ __forceinline__ void edge_sd(long long e, const int* __restrict__ src,
                                        const int* __restrict__ dst, int& s, int& d) {
    if (e < NEDGES) { s = src[e]; d = dst[e]; }
    else            { s = (int)(e - NEDGES); d = s; }
}

// ---- pack W[K x NOUT] (row major f32) into WMMA-B bf16 fragments ----
// layout: Wp[((ct*(K/32)+ks)*32 + lane)*16 + e], K index = ks*32 + 8*(lane>>4) + (e&7) + 16*(e>>3),
//         N index = ct*16 + (lane&15)
__global__ void pack_w_kernel(const float* __restrict__ W, unsigned short* __restrict__ Wp,
                              int K, int NOUT) {
    int idx = blockIdx.x * 256 + threadIdx.x;
    int KS = K >> 5;
    int total = (NOUT >> 4) * KS * 512;
    if (idx >= total) return;
    int e    = idx & 15;
    int lane = (idx >> 4) & 31;
    int rem  = idx % (KS * 512);
    int ks   = rem >> 9;
    int ct   = idx / (KS * 512);
    int n    = ct * 16 + (lane & 15);
    int kk   = ks * 32 + 8 * (lane >> 4) + (e & 7) + ((e >> 3) << 4);
    Wp[idx] = f2bf(W[kk * NOUT + n]);
}

// ---- WMMA GEMM: Out[N x NOUT] = X[N x K](f32 -> bf16) @ Wp ----
// block = 128 threads (4 waves); each wave owns one 16x16 tile; grid.y covers NOUT/64
template<int K, int NOUT>
__global__ __launch_bounds__(128) void wmma_gemm_kernel(const float* __restrict__ X,
                                                        const unsigned short* __restrict__ Wp,
                                                        float* __restrict__ Out) {
    __shared__ unsigned short As[16 * K];      // bf16 A tile, row major [16][K]
    const int rowBase = blockIdx.x * 16;
    const int tid = threadIdx.x;
    for (int i = tid; i < 16 * K; i += 128) {
        int r = i / K, k = i % K;
        As[i] = f2bf(X[(long long)(rowBase + r) * K + k]);
    }
    __syncthreads();

    const int wave = tid >> 5;
    const int lane = tid & 31;
    const int ct   = blockIdx.y * 4 + wave;    // 16-col tile index
    const int half = lane >> 4;
    const int row  = lane & 15;
    constexpr int KS = K / 32;

    v8f acc = {};
    for (int ks = 0; ks < KS; ++ks) {
        BF16x16 a, b;
        const unsigned short* ap = &As[row * K + ks * 32 + 8 * half];
        #pragma unroll
        for (int v = 0; v < 4; ++v) {          // A: e<8 -> K=e+8h ; e>=8 -> K=e+8+8h
            a.w[v]     = *(const unsigned int*)&ap[2 * v];
            a.w[4 + v] = *(const unsigned int*)&ap[16 + 2 * v];
        }
        b.v = *(const v16bf*)&Wp[((ct * KS + ks) * 32 + lane) * 16];
        acc = __builtin_amdgcn_wmma_f32_16x16x32_bf16(
                  false, a.v, false, b.v, (short)0, acc, false, false);
    }
    const int col = ct * 16 + (lane & 15);
    #pragma unroll
    for (int r = 0; r < 8; ++r)
        Out[(long long)(rowBase + r + 8 * half) * NOUT + col] = acc[r];
}

// ---- attention logits: als/ald [N*H] = sum_c h[(n*H+h)*64+c] * a[h*64+c] ----
__global__ void att_logits_kernel(const float* __restrict__ h, const float* __restrict__ a_s,
                                  const float* __restrict__ a_d, float* __restrict__ als,
                                  float* __restrict__ ald, int HH) {
    int i = blockIdx.x * 256 + threadIdx.x;
    if (i >= NNODES * HH) return;
    int hd = i % HH;
    const float4* hp = (const float4*)(h + (long long)i * CC);
    const float4* as4 = (const float4*)(a_s + hd * CC);
    const float4* ad4 = (const float4*)(a_d + hd * CC);
    float ss = 0.f, sd = 0.f;
    #pragma unroll 4
    for (int c = 0; c < CC / 4; ++c) {
        float4 v = hp[c], a = as4[c], b = ad4[c];
        ss += v.x * a.x + v.y * a.y + v.z * a.z + v.w * a.w;
        sd += v.x * b.x + v.y * b.y + v.z * b.z + v.w * b.w;
    }
    als[i] = ss; ald[i] = sd;
}

// ---- edge pass 1: leaky-relu logit, store, atomicMax per (dst, head) ----
__global__ void edge_max_kernel(const int* __restrict__ src, const int* __restrict__ dst,
                                const float* __restrict__ als, const float* __restrict__ ald,
                                float* __restrict__ logitbuf, unsigned int* __restrict__ mkey,
                                int HH) {
    long long i = (long long)blockIdx.x * 256 + threadIdx.x;
    if (i >= (long long)EPRIME * HH) return;
    int hd = (int)(i % HH);
    long long e = i / HH;
    int s, d; edge_sd(e, src, dst, s, d);
    float x = als[(long long)s * HH + hd] + ald[(long long)d * HH + hd];
    x = x > 0.f ? x : NEG_SLOPE * x;
    logitbuf[i] = x;
    atomicMax(&mkey[d * HH + hd], fkey(x));
}

// ---- edge pass 2: exp(logit - max), store, atomicAdd Z ----
__global__ void edge_exp_kernel(const int* __restrict__ src, const int* __restrict__ dst,
                                const unsigned int* __restrict__ mkey,
                                float* __restrict__ logitbuf, float* __restrict__ zsum, int HH) {
    long long i = (long long)blockIdx.x * 256 + threadIdx.x;
    if (i >= (long long)EPRIME * HH) return;
    int hd = (int)(i % HH);
    long long e = i / HH;
    int s, d; edge_sd(e, src, dst, s, d);
    float m = fkeyinv(mkey[d * HH + hd]);
    float ex = __expf(logitbuf[i] - m);
    logitbuf[i] = ex;
    atomicAdd(&zsum[d * HH + hd], ex);
}

// ---- edge pass 3: out[d] += alpha * feat[s] ----
// one thread per (edge, head, 16-channel block): alpha computed once,
// 4x float4 gathers (global_load_b128, L2-resident), 16 pipelined atomics
__global__ void edge_scatter_kernel(const int* __restrict__ src, const int* __restrict__ dst,
                                    const float* __restrict__ expb, const float* __restrict__ zsum,
                                    const float* __restrict__ feat, float* __restrict__ outm,
                                    int HH) {
    long long i = (long long)blockIdx.x * 256 + threadIdx.x;
    if (i >= (long long)EPRIME * HH * 4) return;
    int cb = (int)(i & 3) * 16;               // channel block base
    long long eh = i >> 2;
    int hd = (int)(eh % HH);
    long long e = eh / HH;
    int s, d; edge_sd(e, src, dst, s, d);
    float alpha = expb[eh] / (zsum[d * HH + hd] + 1e-16f);
    const float4* fp = (const float4*)(feat + ((long long)s * HH + hd) * CC + cb);
    float* op = outm + ((long long)d * HH + hd) * CC + cb;
    #pragma unroll
    for (int q = 0; q < 4; ++q) {
        float4 v = fp[q];
        atomicAdd(&op[q * 4 + 0], alpha * v.x);
        atomicAdd(&op[q * 4 + 1], alpha * v.y);
        atomicAdd(&op[q * 4 + 2], alpha * v.z);
        atomicAdd(&op[q * 4 + 3], alpha * v.w);
    }
}

// ---- BatchNorm stats (over h+bias), 256 channels ----
__global__ void bn_stats_kernel(const float* __restrict__ h, const float* __restrict__ b1,
                                float* __restrict__ sum, float* __restrict__ sumsq) {
    int c = threadIdx.x;               // 256 channels
    int r0 = blockIdx.x * 50;          // 1000 blocks * 50 rows
    float bb = b1[c];
    float s = 0.f, ss = 0.f;
    for (int r = r0; r < r0 + 50; ++r) {
        float v = h[(long long)r * 256 + c] + bb;
        s += v; ss += v * v;
    }
    atomicAdd(&sum[c], s);
    atomicAdd(&sumsq[c], ss);
}

__global__ void bn_finalize_kernel(const float* __restrict__ sum, const float* __restrict__ sumsq,
                                   const float* __restrict__ gamma,
                                   float* __restrict__ mu, float* __restrict__ rsg) {
    int c = threadIdx.x;
    float m = sum[c] / (float)NNODES;
    float var = sumsq[c] / (float)NNODES - m * m;
    mu[c] = m;
    rsg[c] = rsqrtf(var + BN_EPS) * gamma[c];
}

// ---- in-place: h = relu((h + b1 - mu) * rsg + beta), float4 ----
__global__ void bn_apply_relu_kernel(float* __restrict__ h, const float* __restrict__ b1,
                                     const float* __restrict__ mu, const float* __restrict__ rsg,
                                     const float* __restrict__ beta) {
    long long i = (long long)blockIdx.x * 256 + threadIdx.x;   // N*256/4 threads
    if (i >= (long long)NNODES * 64) return;
    int c = (int)((i * 4) & 255);
    float4 v = ((float4*)h)[i];
    float r0 = fmaxf((v.x + b1[c + 0] - mu[c + 0]) * rsg[c + 0] + beta[c + 0], 0.f);
    float r1 = fmaxf((v.y + b1[c + 1] - mu[c + 1]) * rsg[c + 1] + beta[c + 1], 0.f);
    float r2 = fmaxf((v.z + b1[c + 2] - mu[c + 2]) * rsg[c + 2] + beta[c + 2], 0.f);
    float r3 = fmaxf((v.w + b1[c + 3] - mu[c + 3]) * rsg[c + 3] + beta[c + 3], 0.f);
    ((float4*)h)[i] = make_float4(r0, r1, r2, r3);
}

// ---- in-place: h = relu(h + b2) (layer 2, 64 channels), float4 ----
__global__ void bias_relu_kernel(float* __restrict__ h, const float* __restrict__ b2) {
    long long i = (long long)blockIdx.x * 256 + threadIdx.x;   // N*64/4 threads
    if (i >= (long long)NNODES * 16) return;
    int c = (int)((i * 4) & (CC - 1));
    float4 v = ((float4*)h)[i];
    v.x = fmaxf(v.x + b2[c + 0], 0.f);
    v.y = fmaxf(v.y + b2[c + 1], 0.f);
    v.z = fmaxf(v.z + b2[c + 2], 0.f);
    v.w = fmaxf(v.w + b2[c + 3], 0.f);
    ((float4*)h)[i] = v;
}

// ---- global mean pool: thread per (node, 16-channel block) ----
__global__ void pool_kernel(const float* __restrict__ h2, const int* __restrict__ batch,
                            float* __restrict__ pool, float* __restrict__ cnt) {
    int i = blockIdx.x * 256 + threadIdx.x;    // N*4 threads
    if (i >= NNODES * 4) return;
    int n = i >> 2, cb = (i & 3) * 16;
    int g = batch[n];
    const float4* hp = (const float4*)(h2 + (long long)n * CC + cb);
    float* pp = pool + g * CC + cb;
    #pragma unroll
    for (int q = 0; q < 4; ++q) {
        float4 v = hp[q];
        atomicAdd(&pp[q * 4 + 0], v.x);
        atomicAdd(&pp[q * 4 + 1], v.y);
        atomicAdd(&pp[q * 4 + 2], v.z);
        atomicAdd(&pp[q * 4 + 3], v.w);
    }
    if (cb == 0) atomicAdd(&cnt[g], 1.f);
}

// ---- classifier: per-graph 64 -> 32 (relu) -> 10 ----
__global__ void classifier_kernel(const float* __restrict__ pool, const float* __restrict__ cnt,
                                  const float* __restrict__ cW1, const float* __restrict__ cb1,
                                  const float* __restrict__ cW2, const float* __restrict__ cb2,
                                  float* __restrict__ out) {
    int g = blockIdx.x * 256 + threadIdx.x;
    if (g >= NG) return;
    float p[CC];
    float inv = 1.f / fmaxf(cnt[g], 1.f);
    for (int c = 0; c < CC; ++c) p[c] = pool[g * CC + c] * inv;
    float z[32];
    for (int j = 0; j < 32; ++j) {
        float s = cb1[j];
        for (int c = 0; c < CC; ++c) s += p[c] * cW1[c * 32 + j];
        z[j] = fmaxf(s, 0.f);
    }
    for (int o = 0; o < NOUTC; ++o) {
        float s = cb2[o];
        for (int j = 0; j < 32; ++j) s += z[j] * cW2[j * NOUTC + o];
        out[g * NOUTC + o] = s;
    }
}

extern "C" void kernel_launch(void* const* d_in, const int* in_sizes, int n_in,
                              void* d_out, int out_size, void* d_ws, size_t ws_size,
                              hipStream_t stream) {
    // inputs (setup_inputs order)
    const float* x      = (const float*)d_in[0];
    const int*   eidx   = (const int*)d_in[1];
    const int*   batch  = (const int*)d_in[2];
    const float* W1     = (const float*)d_in[3];
    const float* a_src1 = (const float*)d_in[4];
    const float* a_dst1 = (const float*)d_in[5];
    const float* b1     = (const float*)d_in[6];
    const float* gamma  = (const float*)d_in[7];
    const float* beta   = (const float*)d_in[8];
    const float* W2     = (const float*)d_in[9];
    const float* a_src2 = (const float*)d_in[10];
    const float* a_dst2 = (const float*)d_in[11];
    const float* b2     = (const float*)d_in[12];
    const float* cW1    = (const float*)d_in[13];
    const float* cb1    = (const float*)d_in[14];
    const float* cW2    = (const float*)d_in[15];
    const float* cb2    = (const float*)d_in[16];
    const int* src = eidx;
    const int* dst = eidx + NEDGES;
    float* out = (float*)d_out;

    // ---- carve workspace: zero-init region first (single memset per call) ----
    uint8_t* base = (uint8_t*)d_ws;
    size_t off = 0;
    auto take = [&](size_t bytes) -> void* {
        void* r = base + off;
        off = (off + bytes + 255) & ~(size_t)255;
        return r;
    };
    float*        msg1   = (float*)take((size_t)NNODES * 256 * 4);
    float*        z1     = (float*)take((size_t)NNODES * NH * 4);
    unsigned int* mkey1  = (unsigned int*)take((size_t)NNODES * NH * 4);
    float*        bn_sum = (float*)take(256 * 4);
    float*        bn_ssq = (float*)take(256 * 4);
    float*        msg2   = (float*)take((size_t)NNODES * CC * 4);
    float*        z2     = (float*)take((size_t)NNODES * 4);
    unsigned int* mkey2  = (unsigned int*)take((size_t)NNODES * 4);
    float*        pool   = (float*)take((size_t)NG * CC * 4);
    float*        cnt    = (float*)take((size_t)NG * 4);
    size_t zeroBytes = off;
    unsigned short* W1p  = (unsigned short*)take((size_t)16 * 4 * 512 * 2);
    unsigned short* W2p  = (unsigned short*)take((size_t)4 * 8 * 512 * 2);
    float*        H1     = (float*)take((size_t)NNODES * 256 * 4);
    float*        als1   = (float*)take((size_t)NNODES * NH * 4);
    float*        ald1   = (float*)take((size_t)NNODES * NH * 4);
    float*        expE1  = (float*)take((size_t)EPRIME * NH * 4);
    float*        H2     = (float*)take((size_t)NNODES * CC * 4);
    float*        als2   = (float*)take((size_t)NNODES * 4);
    float*        ald2   = (float*)take((size_t)NNODES * 4);
    float*        expE2  = (float*)take((size_t)EPRIME * 4);
    float*        bn_mu  = (float*)take(256 * 4);
    float*        bn_rsg = (float*)take(256 * 4);
    (void)ws_size; (void)n_in; (void)in_sizes; (void)out_size;

    hipMemsetAsync(d_ws, 0, zeroBytes, stream);

    // weight packing for WMMA B layout
    pack_w_kernel<<<128, 256, 0, stream>>>(W1, W1p, DIN, 256);
    pack_w_kernel<<<64, 256, 0, stream>>>(W2, W2p, 256, CC);

    // ===== layer 1: GATConv(128 -> 4x64, concat) =====
    wmma_gemm_kernel<DIN, 256><<<dim3(NNODES / 16, 4), 128, 0, stream>>>(x, W1p, H1);
    att_logits_kernel<<<(NNODES * NH + 255) / 256, 256, 0, stream>>>(H1, a_src1, a_dst1, als1, ald1, NH);
    {
        int eb = (int)(((long long)EPRIME * NH + 255) / 256);
        edge_max_kernel<<<eb, 256, 0, stream>>>(src, dst, als1, ald1, expE1, mkey1, NH);
        edge_exp_kernel<<<eb, 256, 0, stream>>>(src, dst, mkey1, expE1, z1, NH);
        int sb = (int)(((long long)EPRIME * NH * 4 + 255) / 256);
        edge_scatter_kernel<<<sb, 256, 0, stream>>>(src, dst, expE1, z1, H1, msg1, NH);
    }
    // BatchNorm(256) + ReLU (bias b1 folded into stats and apply)
    bn_stats_kernel<<<1000, 256, 0, stream>>>(msg1, b1, bn_sum, bn_ssq);
    bn_finalize_kernel<<<1, 256, 0, stream>>>(bn_sum, bn_ssq, gamma, bn_mu, bn_rsg);
    bn_apply_relu_kernel<<<(NNODES * 64) / 256, 256, 0, stream>>>(msg1, b1, bn_mu, bn_rsg, beta);

    // ===== layer 2: GATConv(256 -> 64, heads=1) =====
    wmma_gemm_kernel<256, CC><<<dim3(NNODES / 16, 1), 128, 0, stream>>>(msg1, W2p, H2);
    att_logits_kernel<<<(NNODES + 255) / 256, 256, 0, stream>>>(H2, a_src2, a_dst2, als2, ald2, 1);
    {
        int eb = (EPRIME + 255) / 256;
        edge_max_kernel<<<eb, 256, 0, stream>>>(src, dst, als2, ald2, expE2, mkey2, 1);
        edge_exp_kernel<<<eb, 256, 0, stream>>>(src, dst, mkey2, expE2, z2, 1);
        int sb = (int)(((long long)EPRIME * 4 + 255) / 256);
        edge_scatter_kernel<<<sb, 256, 0, stream>>>(src, dst, expE2, z2, H2, msg2, 1);
    }
    bias_relu_kernel<<<(NNODES * 16) / 256, 256, 0, stream>>>(msg2, b2);

    // ===== pooling + classifier =====
    pool_kernel<<<(NNODES * 4 + 255) / 256, 256, 0, stream>>>(msg2, batch, pool, cnt);
    classifier_kernel<<<(NG + 255) / 256, 256, 0, stream>>>(pool, cnt, cW1, cb1, cW2, cb2, out);
}
